// CnfProcessingBlock_15040975470804
// MI455X (gfx1250) — matrix operand order repaired and checked by reading, not compiled, not run on hardware
//
#include <hip/hip_runtime.h>
#include <hip/hip_bf16.h>
#include <cstdint>

typedef __attribute__((ext_vector_type(16))) _Float16 v16h;
typedef __attribute__((ext_vector_type(8)))  float    v8f;

#define C_DIM  128
#define ED_DIM 16
#define W_ELEMS (3 * C_DIM * C_DIM)   // 49152 per hi/lo buffer

__device__ __forceinline__ void atomic_max_f32(float* addr, float val) {
  // Sign-split trick: int max for non-negative, uint min for negative.
  if (val >= 0.0f) atomicMax((int*)addr, __float_as_int(val));
  else             atomicMin((unsigned int*)addr, __float_as_uint(val));
}

// ---------------------------------------------------------------------------
// K0: precompute W^T as f16 hi/lo pairs, once per call.
// WT[t][col][k] layout so a B fragment is 16 contiguous halves per lane.
// ---------------------------------------------------------------------------
__global__ void prep_w_kernel(
    const float* __restrict__ W0, const float* __restrict__ W1,
    const float* __restrict__ W2,
    _Float16* __restrict__ WThi, _Float16* __restrict__ WTlo)
{
  const int idx = blockIdx.x * 256 + threadIdx.x;
  if (idx >= W_ELEMS) return;
  const int t   = idx >> 14;          // variant
  const int rem = idx & 16383;
  const int col = rem >> 7;
  const int k   = rem & 127;
  const float* __restrict__ W = (t == 0) ? W0 : ((t == 1) ? W1 : W2);
  const float x = W[k * C_DIM + col];
  const _Float16 hi = (_Float16)x;
  WThi[idx] = hi;
  WTlo[idx] = (_Float16)(x - (float)hi);
}

// ---------------------------------------------------------------------------
// K1: xs_t = h @ W_t  (3 variants) via f16 hi/lo split WMMA (near-fp32).
// block = (32,8): 8 waves; each wave owns 16 rows x all 128 cols
// (8 v8f accumulators). A fragments converted once per K-chunk and reused
// across 8 column tiles; B fragments are single aligned v16h loads from the
// precomputed transposed f16 weights.
// grid = ((N+127)/128, 1, 3)
// ---------------------------------------------------------------------------
__global__ __launch_bounds__(256) void gemm_xs_kernel(
    const float* __restrict__ h,
    const _Float16* __restrict__ WThi, const _Float16* __restrict__ WTlo,
    float* __restrict__ xs, int N)
{
  const int lane = threadIdx.x;                 // 0..31
  const int wave = threadIdx.y;                 // 0..7
  const int var  = blockIdx.z;

  const int row0 = (blockIdx.x * 8 + wave) * 16;
  if (row0 >= N) return;
  const int l15  = lane & 15;
  const int hsel = lane >> 4;                   // half-wave select

  int arow = row0 + l15;
  if (arow >= N) arow = N - 1;                  // safe read; stores guarded
  const float* __restrict__ hrow = h + (size_t)arow * C_DIM;
  const _Float16* __restrict__ WH = WThi + (size_t)var * C_DIM * C_DIM;
  const _Float16* __restrict__ WL = WTlo + (size_t)var * C_DIM * C_DIM;

  v8f acc[8] = {};
  #pragma unroll
  for (int k0 = 0; k0 < C_DIM; k0 += 32) {
    // A fragment, contiguous k order used here: lanes 0-15 -> K = k0+0..15,
    //                            lanes 16-31 -> K = k0+16..31
    v16h ahi, alo;
    const int ka = k0 + hsel * 16;
    #pragma unroll
    for (int e = 0; e < 8; ++e) {
      float x0 = hrow[ka + e];
      _Float16 h0 = (_Float16)x0;
      ahi[e]     = h0;  alo[e]     = (_Float16)(x0 - (float)h0);
      float x1 = hrow[ka + 8 + e];
      _Float16 h1 = (_Float16)x1;
      ahi[8 + e] = h1;  alo[8 + e] = (_Float16)(x1 - (float)h1);
    }
    // B fragment, same contiguous k order as A: lane holds col N; lanes 0-15 K=k0+0..15,
    // lanes 16-31 K=k0+16..31  ->  16 contiguous halves in WT[col][k]
    const int kb = k0 + hsel * 16;
    #pragma unroll
    for (int ct = 0; ct < 8; ++ct) {
      const size_t boff = (size_t)(ct * 16 + l15) * C_DIM + kb;
      const v16h bhi = *(const v16h*)(WH + boff);
      const v16h blo = *(const v16h*)(WL + boff);
      acc[ct] = __builtin_amdgcn_wmma_f32_16x16x32_f16(false, ahi, false, bhi, (short)0, acc[ct], false, false);
      acc[ct] = __builtin_amdgcn_wmma_f32_16x16x32_f16(false, ahi, false, blo, (short)0, acc[ct], false, false);
      acc[ct] = __builtin_amdgcn_wmma_f32_16x16x32_f16(false, alo, false, bhi, (short)0, acc[ct], false, false);
    }
  }

  // C/D 16x16 f32 layout: element r, lanes 0-15 -> (M=r, N=lane);
  //                       lanes 16-31 -> (M=8+r, N=lane-16)
  float* __restrict__ out = xs + (size_t)var * N * C_DIM;
  #pragma unroll
  for (int ct = 0; ct < 8; ++ct) {
    #pragma unroll
    for (int r = 0; r < 8; ++r) {
      const int orow = row0 + r + hsel * 8;
      if (orow < N) out[(size_t)orow * C_DIM + ct * 16 + l15] = acc[ct][r];
    }
  }
}

// ---------------------------------------------------------------------------
// K2: al_s[t][n] = xs_t[n] . a_src_t ; al_d[t][n] = xs_t[n] . a_dst_t
// One wave per (node, variant). block=256 (8 waves), grid=((N+7)/8, 3)
// ---------------------------------------------------------------------------
__global__ __launch_bounds__(256) void node_scores_kernel(
    const float* __restrict__ xs,
    const float* __restrict__ as0, const float* __restrict__ as1, const float* __restrict__ as2,
    const float* __restrict__ ad0, const float* __restrict__ ad1, const float* __restrict__ ad2,
    float* __restrict__ al_s, float* __restrict__ al_d, int N)
{
  const int lane = threadIdx.x & 31;
  const int wid  = threadIdx.x >> 5;
  const int node = blockIdx.x * 8 + wid;
  const int var  = blockIdx.y;
  if (node >= N) return;

  const float* __restrict__ as = (var == 0) ? as0 : ((var == 1) ? as1 : as2);
  const float* __restrict__ ad = (var == 0) ? ad0 : ((var == 1) ? ad1 : ad2);
  const float4* __restrict__ xr = (const float4*)(xs + ((size_t)var * N + node) * C_DIM);

  float4 v  = xr[lane];
  float4 vs = ((const float4*)as)[lane];
  float4 vd = ((const float4*)ad)[lane];
  float ps = v.x * vs.x + v.y * vs.y + v.z * vs.z + v.w * vs.w;
  float pd = v.x * vd.x + v.y * vd.y + v.z * vd.z + v.w * vd.w;
  #pragma unroll
  for (int off = 16; off > 0; off >>= 1) {
    ps += __shfl_down(ps, off, 32);
    pd += __shfl_down(pd, off, 32);
  }
  if (lane == 0) {
    al_s[(size_t)var * N + node] = ps;
    al_d[(size_t)var * N + node] = pd;
  }
}

// ---------------------------------------------------------------------------
// K3: c_t = We_t @ a_edge_t  (3 x 16 coefficients). One block, 64 threads.
// ---------------------------------------------------------------------------
__global__ void edge_coeff_kernel(
    const float* __restrict__ We0, const float* __restrict__ ae0,
    const float* __restrict__ We1, const float* __restrict__ ae1,
    const float* __restrict__ We2, const float* __restrict__ ae2,
    float* __restrict__ cbuf)
{
  const int t = threadIdx.x >> 4;
  const int d = threadIdx.x & 15;
  if (t >= 3) return;
  const float* __restrict__ We = (t == 0) ? We0 : ((t == 1) ? We1 : We2);
  const float* __restrict__ ae = (t == 0) ? ae0 : ((t == 1) ? ae1 : ae2);
  float s = 0.0f;
  #pragma unroll 8
  for (int j = 0; j < C_DIM; ++j) s += We[d * C_DIM + j] * ae[j];
  cbuf[t * 16 + d] = s;
}

// ---------------------------------------------------------------------------
// K4: init d_out[n,c] = h[n,c] + b_t[c] (residual+bias), m[n]=-inf, s[n]=0
// ---------------------------------------------------------------------------
__global__ void init_kernel(
    const float* __restrict__ h, const int* __restrict__ ntype,
    const float* __restrict__ b0, const float* __restrict__ b1,
    const float* __restrict__ b2,
    float* __restrict__ out, float* __restrict__ m, float* __restrict__ s, int N)
{
  const long long idx = (long long)blockIdx.x * blockDim.x + threadIdx.x;
  if (idx >= (long long)N * C_DIM) return;
  const int n = (int)(idx >> 7);
  const int c = (int)(idx & 127);
  const int t = ntype[n];
  const float* __restrict__ b = (t == 0) ? b0 : ((t == 1) ? b1 : b2);
  out[idx] = h[idx] + b[c];
  if (c == 0) { m[n] = -__builtin_huge_valf(); s[n] = 0.0f; }
}

// ---------------------------------------------------------------------------
// K5: per-edge attention logit + leaky relu + segment max (atomic)
// ---------------------------------------------------------------------------
__global__ void edge_logit_kernel(
    const int* __restrict__ eidx, const int* __restrict__ ntype,
    const float* __restrict__ al_s, const float* __restrict__ al_d,
    const float* __restrict__ eattr, const float* __restrict__ cbuf,
    float* __restrict__ score, float* __restrict__ m, int N, int E)
{
  const int e = blockIdx.x * blockDim.x + threadIdx.x;
  if (e >= E) return;
  const int src = eidx[e];
  const int dst = eidx[E + e];
  const int t   = ntype[dst];
  float a = al_s[(size_t)t * N + src] + al_d[(size_t)t * N + dst];
  const float* __restrict__ ea = eattr + (size_t)e * ED_DIM;
  const float* __restrict__ ct = cbuf + t * 16;
  float ae = 0.0f;
  #pragma unroll
  for (int j = 0; j < ED_DIM; ++j) ae += ea[j] * ct[j];
  a += ae;
  a = (a > 0.0f) ? a : 0.2f * a;        // leaky_relu(0.2)
  score[e] = a;
  atomic_max_f32(&m[dst], a);
}

// ---------------------------------------------------------------------------
// K6: e = exp(a - m[dst]); s[dst] += e  (atomic); score <- e
// ---------------------------------------------------------------------------
__global__ void edge_exp_kernel(
    const int* __restrict__ eidx, const float* __restrict__ m,
    float* __restrict__ score, float* __restrict__ s, int E)
{
  const int e = blockIdx.x * blockDim.x + threadIdx.x;
  if (e >= E) return;
  const int dst = eidx[E + e];
  const float ev = expf(score[e] - m[dst]);
  score[e] = ev;
  atomicAdd(&s[dst], ev);
}

// ---------------------------------------------------------------------------
// K7: out[dst] += (e / (s[dst]+eps)) * xs_t[src].  One wave per edge,
// lane handles 4 channels (float4 gather + 4 float atomic adds).
// ---------------------------------------------------------------------------
__global__ __launch_bounds__(256) void aggregate_kernel(
    const int* __restrict__ eidx, const int* __restrict__ ntype,
    const float* __restrict__ xs, const float* __restrict__ score,
    const float* __restrict__ s, float* __restrict__ out, int N, int E)
{
  const int lane = threadIdx.x & 31;
  const int wid  = threadIdx.x >> 5;
  const int e    = blockIdx.x * 8 + wid;
  if (e >= E) return;
  const int src = eidx[e];
  const int dst = eidx[E + e];
  const int t   = ntype[dst];
  const float w = score[e] / (s[dst] + 1e-16f);

  const float4* __restrict__ xr = (const float4*)(xs + ((size_t)t * N + src) * C_DIM);
  const float4 v = xr[lane];
  float* __restrict__ o = out + (size_t)dst * C_DIM + lane * 4;
  atomicAdd(o + 0, w * v.x);
  atomicAdd(o + 1, w * v.y);
  atomicAdd(o + 2, w * v.z);
  atomicAdd(o + 3, w * v.w);
}

// ---------------------------------------------------------------------------
// K8: in-place ReLU on d_out
// ---------------------------------------------------------------------------
__global__ void relu_kernel(float* __restrict__ out, long long n)
{
  const long long idx = (long long)blockIdx.x * blockDim.x + threadIdx.x;
  if (idx < n) out[idx] = fmaxf(out[idx], 0.0f);
}

// ---------------------------------------------------------------------------
extern "C" void kernel_launch(void* const* d_in, const int* in_sizes, int n_in,
                              void* d_out, int out_size, void* d_ws, size_t ws_size,
                              hipStream_t stream) {
  const float* h     = (const float*)d_in[0];
  const int*   eidx  = (const int*)  d_in[1];   // [2, E]
  const float* eattr = (const float*)d_in[2];   // [E, 16]
  const int*   ntype = (const int*)  d_in[3];   // [N]
  // per-variant params: W, a_src, a_dst, We, a_edge, b
  const float* Wv  = (const float*)d_in[4];
  const float* asv = (const float*)d_in[5];
  const float* adv = (const float*)d_in[6];
  const float* Wev = (const float*)d_in[7];
  const float* aev = (const float*)d_in[8];
  const float* bv  = (const float*)d_in[9];
  const float* Wr  = (const float*)d_in[10];
  const float* asr = (const float*)d_in[11];
  const float* adr = (const float*)d_in[12];
  const float* Wer = (const float*)d_in[13];
  const float* aer = (const float*)d_in[14];
  const float* br  = (const float*)d_in[15];
  const float* Wi  = (const float*)d_in[16];
  const float* asi = (const float*)d_in[17];
  const float* adi = (const float*)d_in[18];
  const float* Wei = (const float*)d_in[19];
  const float* aei = (const float*)d_in[20];
  const float* bi  = (const float*)d_in[21];

  const int N = in_sizes[3];
  const int E = in_sizes[2] / ED_DIM;
  float* out = (float*)d_out;

  // workspace layout (f16 weight buffers first -> everything stays aligned)
  _Float16* WThi = (_Float16*)d_ws;                 // 49152 halves
  _Float16* WTlo = WThi + W_ELEMS;                  // 49152 halves
  float* xs    = (float*)(WTlo + W_ELEMS);          // 3*N*128
  float* al_s  = xs   + (size_t)3 * N * C_DIM;      // 3*N
  float* al_d  = al_s + (size_t)3 * N;              // 3*N
  float* mbuf  = al_d + (size_t)3 * N;              // N
  float* sbuf  = mbuf + N;                          // N
  float* score = sbuf + N;                          // E
  float* cbuf  = score + E;                         // 48

  // K0: f16 hi/lo transposed weights (once per call; tiny)
  prep_w_kernel<<<(W_ELEMS + 255) / 256, 256, 0, stream>>>(Wv, Wr, Wi, WThi, WTlo);

  // K1: WMMA GEMM  xs_t = h @ W_t
  {
    dim3 grid((N + 127) / 128, 1, 3), block(32, 8, 1);
    gemm_xs_kernel<<<grid, block, 0, stream>>>(h, WThi, WTlo, xs, N);
  }
  // K2: per-node attention projections
  {
    dim3 grid((N + 7) / 8, 3), block(256);
    node_scores_kernel<<<grid, block, 0, stream>>>(xs, asv, asr, asi,
                                                   adv, adr, adi, al_s, al_d, N);
  }
  // K3: edge-attr coefficients
  edge_coeff_kernel<<<1, 64, 0, stream>>>(Wev, aev, Wer, aer, Wei, aei, cbuf);

  // K4: init out = h + b_t ; m = -inf ; s = 0
  {
    const long long tot = (long long)N * C_DIM;
    init_kernel<<<(unsigned)((tot + 255) / 256), 256, 0, stream>>>(
        h, ntype, bv, br, bi, out, mbuf, sbuf, N);
  }
  // K5: logits + leaky relu + segment max
  edge_logit_kernel<<<(E + 255) / 256, 256, 0, stream>>>(
      eidx, ntype, al_s, al_d, eattr, cbuf, score, mbuf, N, E);
  // K6: exp + segment sum
  edge_exp_kernel<<<(E + 255) / 256, 256, 0, stream>>>(eidx, mbuf, score, sbuf, E);
  // K7: weighted aggregation (wave per edge)
  aggregate_kernel<<<(E + 7) / 8, 256, 0, stream>>>(
      eidx, ntype, xs, score, sbuf, out, N, E);
  // K8: ReLU
  {
    const long long tot = (long long)N * C_DIM;
    relu_kernel<<<(unsigned)((tot + 255) / 256), 256, 0, stream>>>(out, tot);
  }
}
